// LSTMLayer_80719615361196
// MI455X (gfx1250) — matrix operand order repaired and checked by reading, not compiled, not run on hardware
//
#include <hip/hip_runtime.h>
#include <hip/hip_bf16.h>
#include <stdint.h>

// Problem sizes (fixed by the reference): B=8, C_in=256, T=4096, C_out=256
#define BQ     8
#define CIN    256
#define TQ     4096
#define NQ     1024          // 4*C_out
#define KQ     512           // 2*C_in
#define SLAB_ELEMS 8448      // (32+1)*256: contiguous A elements covering a 32-row M tile, all K

typedef __bf16 bf16_t;
typedef bf16_t v16bf __attribute__((ext_vector_type(16)));
typedef bf16_t v8bf  __attribute__((ext_vector_type(8)));
typedef float  v8f   __attribute__((ext_vector_type(8)));
typedef unsigned int uint32x4 __attribute__((ext_vector_type(4)));
typedef int          int32x8  __attribute__((ext_vector_type(8)));
typedef int          int32x4  __attribute__((ext_vector_type(4)));

#if __has_builtin(__builtin_amdgcn_tensor_load_to_lds)
#define HAVE_TDM 1
#else
#define HAVE_TDM 0
#endif

// ---------------------------------------------------------------------------
// Kernel 1: transpose + causal-pad + fp32->bf16 hi/lo split.
// Xpad[b, t', c] for t' in [0, T]:  t'==0 -> 0 else x[b, c, t'-1].
// A GEMM row for output time t is then the 512 contiguous bf16 at Xpad[b,t,0].
// ---------------------------------------------------------------------------
__global__ void qrnn_pack_x(const float* __restrict__ x,
                            bf16_t* __restrict__ Ahi, bf16_t* __restrict__ Alo) {
    __shared__ float tile[32][33];
    const int b  = blockIdx.z;
    const int t0 = blockIdx.x * 32;   // over t' in [0, T]
    const int c0 = blockIdx.y * 32;
    const int tx = threadIdx.x, ty = threadIdx.y;

    const int tp = t0 + tx;
    float v = 0.0f;
    if (tp >= 1 && tp <= TQ)
        v = x[((size_t)b * CIN + (c0 + ty)) * TQ + (tp - 1)];
    tile[ty][tx] = v;
    __syncthreads();

    const int tpw = t0 + ty;
    if (tpw <= TQ) {
        float val = tile[tx][ty];
        bf16_t hi = (bf16_t)val;
        bf16_t lo = (bf16_t)(val - (float)hi);
        size_t idx = ((size_t)b * (TQ + 1) + tpw) * CIN + c0 + tx;
        Ahi[idx] = hi;
        Alo[idx] = lo;
    }
}

// ---------------------------------------------------------------------------
// Kernel 2: pack Conv1d weight [4*Cout, Cin, 2] into Wt[n, k] bf16 hi/lo.
// ---------------------------------------------------------------------------
__global__ void qrnn_pack_w(const float* __restrict__ W,
                            bf16_t* __restrict__ Whi, bf16_t* __restrict__ Wlo) {
    const int idx = blockIdx.x * 256 + threadIdx.x;   // n*512 + k
    const int n = idx >> 9;
    const int k = idx & 511;
    float v = (k < 256) ? W[(size_t)n * 512 + k * 2 + 0]
                        : W[(size_t)n * 512 + (k - 256) * 2 + 1];
    bf16_t hi = (bf16_t)v;
    bf16_t lo = (bf16_t)(v - (float)hi);
    Whi[idx] = hi;
    Wlo[idx] = lo;
}

// ---------------------------------------------------------------------------
// TDM helper: 1-row tile of `nelems` 2-byte elements, contiguous in memory,
// into LDS at lds_addr. D# packing per CDNA5 ISA.
// This toolchain uses the 6-arg builtin: (g0, g1, g2, g3, g4, cpol).
// ---------------------------------------------------------------------------
__device__ __forceinline__ void tdm_load_1d(uint64_t gaddr, uint32_t lds_addr,
                                            uint32_t nelems) {
#if HAVE_TDM
    uint32x4 g0;
    g0[0] = 1u;                                            // count=1 (valid user D#)
    g0[1] = lds_addr;                                      // lds_addr (bytes)
    g0[2] = (uint32_t)gaddr;                               // global_addr[31:0]
    g0[3] = (uint32_t)((gaddr >> 32) & 0x01FFFFFFu)        // global_addr[56:32]
          | (2u << 30);                                    // type=2 ("image")
    int32x8 g1;
    g1[0] = (int)(1u << 16);                               // data_size=1 (2 bytes)
    g1[1] = (int)((nelems & 0xFFFFu) << 16);               // tensor_dim0[15:0]
    g1[2] = (int)(((nelems >> 16) & 0xFFFFu) | (1u << 16));// dim0[31:16] | tensor_dim1=1
    g1[3] = (int)((nelems & 0xFFFFu) << 16);               // dim1[31:16]=0 | tile_dim0
    g1[4] = 1;                                             // tile_dim1=1, tile_dim2=0
    g1[5] = (int)nelems;                                   // tensor_dim0_stride[31:0]
    g1[6] = 0;                                             // stride0[47:32] | stride1 lo
    g1[7] = 0;
    int32x4 z4 = {0, 0, 0, 0};
    int32x8 z8 = {0, 0, 0, 0, 0, 0, 0, 0};
    __builtin_amdgcn_tensor_load_to_lds(g0, g1, z4, z4, z8, 0);
#else
    (void)gaddr; (void)lds_addr; (void)nelems;
#endif
}

// ---------------------------------------------------------------------------
// Kernel 3: GEMM via v_wmma_f32_16x16x32_bf16 with bf16 hi/lo fp32 emulation:
//   acc += Ahi*Bhi + Ahi*Blo + Alo*Bhi
// Block = 4 waves; each wave owns a 32(t) x 32(n) tile (4 accumulators, 12
// WMMAs per K-step). A slab (32 rows x all K) is one contiguous 8448-elem run
// staged into LDS by the Tensor Data Mover; B streams from global (L2-resident).
// Fused epilogue: +bias, tanh (Z) / sigmoid (F,O,I), store gate-major G[b,n,t].
// ---------------------------------------------------------------------------
__device__ __forceinline__ v16bf afrag(const bf16_t* rowBase, int kk, int h) {
    v8bf c0 = *(const v8bf*)(rowBase + kk + 8 * h);
    v8bf c1 = *(const v8bf*)(rowBase + kk + 16 + 8 * h);
    return __builtin_shufflevector(c0, c1, 0,1,2,3,4,5,6,7,8,9,10,11,12,13,14,15);
}

__device__ __forceinline__ void store_tile(v8f acc, float bv, int gate, float* gp) {
    float vals[8];
#pragma unroll
    for (int i = 0; i < 8; ++i) {
        float v = acc[i] + bv;
        vals[i] = (gate == 0) ? tanhf(v) : 1.0f / (1.0f + __expf(-v));
    }
    *(float4*)(gp)     = make_float4(vals[0], vals[1], vals[2], vals[3]);
    *(float4*)(gp + 4) = make_float4(vals[4], vals[5], vals[6], vals[7]);
}

__global__ void qrnn_gemm(const bf16_t* __restrict__ Ahi, const bf16_t* __restrict__ Alo,
                          const bf16_t* __restrict__ Whi, const bf16_t* __restrict__ Wlo,
                          const float* __restrict__ bias, float* __restrict__ G) {
    __shared__ bf16_t ldsA[2 * SLAB_ELEMS];               // hi slab then lo slab
    const int lane = threadIdx.x & 31;
    const int wv   = threadIdx.x >> 5;                    // 0..3
    const int t0   = blockIdx.x * 32;
    const int nblk = blockIdx.y * 128;
    const int b    = blockIdx.z;
    const int l15  = lane & 15;                           // A row (M) / B column (N)
    const int h    = lane >> 4;

    const size_t slabOff = ((size_t)b * (TQ + 1) + t0) * CIN;
#if HAVE_TDM
    if (wv == 0) {
        uint32_t lds0 = (uint32_t)(uintptr_t)(&ldsA[0]);
        tdm_load_1d((uint64_t)(uintptr_t)(Ahi + slabOff), lds0, SLAB_ELEMS);
        tdm_load_1d((uint64_t)(uintptr_t)(Alo + slabOff),
                    lds0 + SLAB_ELEMS * (uint32_t)sizeof(bf16_t), SLAB_ELEMS);
        __builtin_amdgcn_s_wait_tensorcnt(0);
    }
    __syncthreads();
#else
    for (int i = threadIdx.x * 8; i < SLAB_ELEMS; i += 128 * 8) {
        *(v8bf*)&ldsA[i]              = *(const v8bf*)(Ahi + slabOff + i);
        *(v8bf*)&ldsA[SLAB_ELEMS + i] = *(const v8bf*)(Alo + slabOff + i);
    }
    __syncthreads();
#endif

    // LDS A row bases: M tile 0 (rows 0..15) and M tile 1 (rows 16..31)
    const bf16_t* a0h = &ldsA[0]          + (size_t)(l15)      * CIN;
    const bf16_t* a1h = &ldsA[0]          + (size_t)(16 + l15) * CIN;
    const bf16_t* a0l = &ldsA[SLAB_ELEMS] + (size_t)(l15)      * CIN;
    const bf16_t* a1l = &ldsA[SLAB_ELEMS] + (size_t)(16 + l15) * CIN;

    // B rows for this wave's two 16-wide n tiles
    const int nbase = nblk + wv * 32;
    const bf16_t* bh0 = Whi + (size_t)(nbase + l15)      * KQ;
    const bf16_t* bh1 = Whi + (size_t)(nbase + 16 + l15) * KQ;
    const bf16_t* bl0 = Wlo + (size_t)(nbase + l15)      * KQ;
    const bf16_t* bl1 = Wlo + (size_t)(nbase + 16 + l15) * KQ;

    v8f acc00 = {0.f,0.f,0.f,0.f,0.f,0.f,0.f,0.f};
    v8f acc01 = acc00, acc10 = acc00, acc11 = acc00;

    for (int kk = 0; kk < KQ; kk += 32) {
        v16bf Ah0 = afrag(a0h, kk, h);
        v16bf Ah1 = afrag(a1h, kk, h);
        v16bf Al0 = afrag(a0l, kk, h);
        v16bf Al1 = afrag(a1l, kk, h);

        // B per-lane K pattern: {kk+8h .. +7} then {kk+16+8h .. +7} (matches A)
        v16bf Bh0 = afrag(bh0, kk, h);
        v16bf Bh1 = afrag(bh1, kk, h);
        v16bf Bl0 = afrag(bl0, kk, h);
        v16bf Bl1 = afrag(bl1, kk, h);

        __builtin_prefetch(bh0 + kk + 64, 0, 3);
        __builtin_prefetch(bh1 + kk + 64, 0, 3);

        acc00 = __builtin_amdgcn_wmma_f32_16x16x32_bf16(false, Ah0, false, Bh0, (short)0, acc00, false, false);
        acc00 = __builtin_amdgcn_wmma_f32_16x16x32_bf16(false, Ah0, false, Bl0, (short)0, acc00, false, false);
        acc00 = __builtin_amdgcn_wmma_f32_16x16x32_bf16(false, Al0, false, Bh0, (short)0, acc00, false, false);

        acc01 = __builtin_amdgcn_wmma_f32_16x16x32_bf16(false, Ah0, false, Bh1, (short)0, acc01, false, false);
        acc01 = __builtin_amdgcn_wmma_f32_16x16x32_bf16(false, Ah0, false, Bl1, (short)0, acc01, false, false);
        acc01 = __builtin_amdgcn_wmma_f32_16x16x32_bf16(false, Al0, false, Bh1, (short)0, acc01, false, false);

        acc10 = __builtin_amdgcn_wmma_f32_16x16x32_bf16(false, Ah1, false, Bh0, (short)0, acc10, false, false);
        acc10 = __builtin_amdgcn_wmma_f32_16x16x32_bf16(false, Ah1, false, Bl0, (short)0, acc10, false, false);
        acc10 = __builtin_amdgcn_wmma_f32_16x16x32_bf16(false, Al1, false, Bh0, (short)0, acc10, false, false);

        acc11 = __builtin_amdgcn_wmma_f32_16x16x32_bf16(false, Ah1, false, Bh1, (short)0, acc11, false, false);
        acc11 = __builtin_amdgcn_wmma_f32_16x16x32_bf16(false, Ah1, false, Bl1, (short)0, acc11, false, false);
        acc11 = __builtin_amdgcn_wmma_f32_16x16x32_bf16(false, Al1, false, Bh1, (short)0, acc11, false, false);
    }

    // Epilogue: D element i -> (m = mt*16 + i + 8h, n = ntile + l15)
    const int n0t = nbase, n1t = nbase + 16;
    const float bv0 = bias[n0t + l15], bv1 = bias[n1t + l15];
    const int g0t = n0t >> 8, g1t = n1t >> 8;             // 0=Z(tanh), else sigmoid

    float* base = G + ((size_t)b * NQ) * TQ + t0 + 8 * h;
    store_tile(acc00, bv0, g0t, base + (size_t)(n0t + l15) * TQ);
    store_tile(acc01, bv1, g1t, base + (size_t)(n1t + l15) * TQ);
    store_tile(acc10, bv0, g0t, base + (size_t)(n0t + l15) * TQ + 16);
    store_tile(acc11, bv1, g1t, base + (size_t)(n1t + l15) * TQ + 16);
}

// ---------------------------------------------------------------------------
// Kernel 4: linear recurrence c_t = f*c_{t-1} + i*z per (b, channel) chain,
// block-parallel scan of affine composites (A,B) over T=4096; h = o * c.
// ---------------------------------------------------------------------------
__global__ void qrnn_scan(const float* __restrict__ G, float* __restrict__ out) {
    const int bc = blockIdx.x;
    const int c = bc & 255;
    const int b = bc >> 8;
    const float* Zp = G + ((size_t)b * NQ + 0   + c) * TQ;
    const float* Fp = G + ((size_t)b * NQ + 256 + c) * TQ;
    const float* Op = G + ((size_t)b * NQ + 512 + c) * TQ;
    const float* Ip = G + ((size_t)b * NQ + 768 + c) * TQ;

    const int tid = threadIdx.x;          // 256 threads, 16 steps each
    const int tb = tid * 16;

    float Ac = 1.0f, Bc = 0.0f;
    float fv[16], izv[16];
#pragma unroll
    for (int j = 0; j < 16; ++j) {
        float f  = Fp[tb + j];
        float iz = Ip[tb + j] * Zp[tb + j];
        fv[j] = f;  izv[j] = iz;
        Ac = f * Ac;
        Bc = f * Bc + iz;
    }

    __shared__ float sA[256], sB[256];
    sA[tid] = Ac; sB[tid] = Bc;
    __syncthreads();
    for (int off = 1; off < 256; off <<= 1) {
        float a2 = sA[tid], b2 = sB[tid];
        float a1 = 1.0f, b1 = 0.0f;
        if (tid >= off) { a1 = sA[tid - off]; b1 = sB[tid - off]; }
        __syncthreads();
        sA[tid] = a2 * a1;
        sB[tid] = a2 * b1 + b2;
        __syncthreads();
    }
    float cc = (tid == 0) ? 0.0f : sB[tid - 1];   // carry into this segment

    float* op = out + ((size_t)b * 256 + c) * TQ + tb;
#pragma unroll
    for (int j = 0; j < 16; ++j) {
        cc = fv[j] * cc + izv[j];
        op[j] = Op[tb + j] * cc;
    }
}

// ---------------------------------------------------------------------------
extern "C" void kernel_launch(void* const* d_in, const int* in_sizes, int n_in,
                              void* d_out, int out_size, void* d_ws, size_t ws_size,
                              hipStream_t stream) {
    const float* x    = (const float*)d_in[0];   // [8, 256, 4096]
    const float* W    = (const float*)d_in[1];   // [1024, 256, 2]
    const float* bias = (const float*)d_in[2];   // [1024]
    float* out = (float*)d_out;                  // [8, 256, 4096]

    const size_t A_ELEMS = (size_t)BQ * (TQ + 1) * CIN;   // 8,390,656
    const size_t W_ELEMS = (size_t)NQ * KQ;               //   524,288
    char* ws = (char*)d_ws;
    bf16_t* Ahi = (bf16_t*)ws;
    bf16_t* Alo = Ahi + A_ELEMS;
    bf16_t* Whi = Alo + A_ELEMS;
    bf16_t* Wlo = Whi + W_ELEMS;
    float*  G   = (float*)(Wlo + W_ELEMS);                // [8, 1024, 4096] f32

    {   // 1) pack x -> causal-padded, transposed, bf16 hi/lo
        dim3 grid((TQ + 1 + 31) / 32, CIN / 32, BQ);      // (129, 8, 8)
        dim3 block(32, 32);
        qrnn_pack_x<<<grid, block, 0, stream>>>(x, Ahi, Alo);
    }
    {   // 2) pack W -> Wt[n,k] bf16 hi/lo
        qrnn_pack_w<<<dim3((NQ * KQ) / 256), dim3(256), 0, stream>>>(W, Whi, Wlo);
    }
    {   // 3) TDM-staged WMMA GEMM + fused activation epilogue -> G[b, n, t]
        dim3 grid(TQ / 32, NQ / 128, BQ);                 // (128, 8, 8)
        qrnn_gemm<<<grid, dim3(128), 0, stream>>>(Ahi, Alo, Whi, Wlo, bias, G);
    }
    {   // 4) parallel associative scan + output gate
        qrnn_scan<<<dim3(BQ * 256), dim3(256), 0, stream>>>(G, out);
    }
}
